// MYMODEL_89421219103600
// MI455X (gfx1250) — matrix operand order repaired and checked by reading, not compiled, not run on hardware
//
#include <hip/hip_runtime.h>
#include <math.h>

typedef __bf16 bf16_t;
typedef __attribute__((ext_vector_type(16))) __bf16 v16bf;
typedef __attribute__((ext_vector_type(8)))  __bf16 v8bf;
typedef __attribute__((ext_vector_type(8)))  float  v8f;

#define MAXK 16

// fast hardware sqrt: single v_sqrt_f32 (~1 ulp), no IEEE refinement sequence
__device__ __forceinline__ float fast_sqrtf(float x) { return __builtin_amdgcn_sqrtf(x); }

// ---------------- fragment loaders (layouts per CDNA5 ISA 7.12.2) ----------------
// A (16x32 bf16): lane L: m=L&15; lanes<16 hold K {0..7, 16..23}, lanes>=16 hold {8..15, 24..31}
__device__ __forceinline__ v16bf load_a_frag(const bf16_t* F, int row0, int k0, int lane, int D) {
    int m  = lane & 15;
    int hi = lane >> 4;
    const bf16_t* p = F + (size_t)(row0 + m) * D + k0 + hi * 8;
    v8bf lo = *(const v8bf*)p;          // K base .. base+7
    v8bf hh = *(const v8bf*)(p + 16);   // K base+16 .. base+23
    return __builtin_shufflevector(lo, hh, 0,1,2,3,4,5,6,7,8,9,10,11,12,13,14,15);
}

// B (32x16 bf16): lane L: n=L&15; lanes<16 hold K {0..7, 16..23}, lanes>=16 hold {8..15, 24..31}
__device__ __forceinline__ v16bf load_b_frag(const bf16_t* F, int col0, int k0, int lane, int D) {
    int n  = lane & 15;
    int hi = lane >> 4;
    const bf16_t* p = F + (size_t)(col0 + n) * D + k0 + hi * 8;
    v8bf lo = *(const v8bf*)p;
    v8bf hh = *(const v8bf*)(p + 16);
    return __builtin_shufflevector(lo, hh, 0,1,2,3,4,5,6,7,8,9,10,11,12,13,14,15);
}

// ---------------- pass 0a: sq norms + bf16 conversion ----------------
__global__ void hg_prep_features(const float* __restrict__ F, bf16_t* __restrict__ Fb,
                                 float* __restrict__ sqn, int D) {
    __shared__ float red[256];
    int row = blockIdx.x;
    const float* fr = F + (size_t)row * D;
    bf16_t* fb = Fb + (size_t)row * D;
    float s = 0.f;
    for (int j = threadIdx.x; j < D; j += 256) {
        float v = fr[j];
        fb[j] = (bf16_t)v;
        s += v * v;
    }
    red[threadIdx.x] = s;
    __syncthreads();
    for (int off = 128; off > 0; off >>= 1) {
        if ((int)threadIdx.x < off) red[threadIdx.x] += red[threadIdx.x + off];
        __syncthreads();
    }
    if (threadIdx.x == 0) sqn[row] = red[0];
}

// ---------------- pass 0b: adjacency row sums ----------------
__global__ void hg_adj_rowsum(const float* __restrict__ adj, float* __restrict__ adjSum, int N) {
    __shared__ float red[256];
    int row = blockIdx.x;
    const float* ar = adj + (size_t)row * N;
    float s = 0.f;
    for (int j = threadIdx.x; j < N; j += 256) s += ar[j];
    red[threadIdx.x] = s;
    __syncthreads();
    for (int off = 128; off > 0; off >>= 1) {
        if ((int)threadIdx.x < off) red[threadIdx.x] += red[threadIdx.x + off];
        __syncthreads();
    }
    if (threadIdx.x == 0) adjSum[row] = red[0];
}

// ---------------- pass 1: WMMA GEMM tile -> eud partial row sums ----------------
// blockIdx.x = column-tile index, blockIdx.y = row-tile index, tile 128x128, 8 waves
// __launch_bounds__(256,1): large VGPR budget so the 2x4 accumulator tile
// plus pipelined A/B fragments never spill to scratch.
__global__ void __launch_bounds__(256, 1)
hg_gemm_eud_partial(const bf16_t* __restrict__ F, const float* __restrict__ sqn,
                    float* __restrict__ eudPartial, int N, int D) {
    __shared__ float partial[128][2];
    int tid  = threadIdx.x;
    int lane = tid & 31;
    int w    = tid >> 5;
    partial[tid >> 1][tid & 1] = 0.f;   // 256 entries by 256 threads
    __syncthreads();

    int tileRow = blockIdx.y * 128;
    int tileCol = blockIdx.x * 128;
    int waveRow = (w & 3) * 32;
    int waveCol = (w >> 2) * 64;

    v8f acc[2][4] = {};
    for (int k0 = 0; k0 < D; k0 += 32) {
        v16bf aF[2], bF[4];
#pragma unroll
        for (int a = 0; a < 2; a++)
            aF[a] = load_a_frag(F, tileRow + waveRow + a * 16, k0, lane, D);
#pragma unroll
        for (int b = 0; b < 4; b++)
            bF[b] = load_b_frag(F, tileCol + waveCol + b * 16, k0, lane, D);
#pragma unroll
        for (int a = 0; a < 2; a++)
#pragma unroll
            for (int b = 0; b < 4; b++)
                acc[a][b] = __builtin_amdgcn_wmma_f32_16x16x32_bf16(
                    false, aF[a], false, bF[b], (short)0, acc[a][b], false, false);
    }

    int hi = lane >> 4;
    int nn = lane & 15;
    float sqnj[4];
#pragma unroll
    for (int b = 0; b < 4; b++) sqnj[b] = sqn[tileCol + waveCol + b * 16 + nn];

#pragma unroll
    for (int a = 0; a < 2; a++) {
#pragma unroll
        for (int r = 0; r < 8; r++) {
            int i = tileRow + waveRow + a * 16 + r + 8 * hi;
            float sqni = sqn[i];
            float s = 0.f;
#pragma unroll
            for (int b = 0; b < 4; b++) {
                int j = tileCol + waveCol + b * 16 + nn;
                float g  = acc[a][b][r];
                float sq = sqni + sqnj[b] - 2.f * g;
                float e  = (sq > 0.f && i != j) ? fast_sqrtf(sq) : 0.f;
                s += e;
            }
            // reduce across the 16-lane half (row is uniform within each half)
#pragma unroll
            for (int off = 1; off < 16; off <<= 1) s += __shfl_xor(s, off, 32);
            if (nn == 0) partial[waveRow + a * 16 + r + 8 * hi][w >> 2] = s;
        }
    }
    __syncthreads();
    if (tid < 128)
        eudPartial[(size_t)blockIdx.x * N + tileRow + tid] = partial[tid][0] + partial[tid][1];
}

// ---------------- pass 1b: reduce partials -> eud row sums (deterministic order) ----------------
__global__ void hg_eud_reduce(const float* __restrict__ part, float* __restrict__ eudSum,
                              int N, int T) {
    int i = blockIdx.x * 256 + threadIdx.x;
    if (i < N) {
        float s = 0.f;
        for (int t = 0; t < T; t++) s += part[(size_t)t * N + i];
        eudSum[i] = s;
    }
}

// ---------------- pass 2: WMMA GEMM recompute + fused distance + per-row top-k ----------------
// block handles 16 rows; 8 waves, wave w owns 32 columns of each 256-col chunk
__global__ void __launch_bounds__(256, 1)
hg_topk(const bf16_t* __restrict__ F, const float* __restrict__ sqn,
        const float* __restrict__ adj, const float* __restrict__ adjSum,
        const float* __restrict__ eudSum, float* __restrict__ out,
        int N, int D, int kk) {
    __shared__ float distLDS[16][256];
    __shared__ float topV[16][MAXK];
    __shared__ int   topJ[16][MAXK];
    __shared__ float rowInvAdj[16], rowInvEud[16], rowSqn[16];

    int tid = threadIdx.x, lane = tid & 31, w = tid >> 5;
    int rowBase = blockIdx.x * 16;

    if (tid < 16) {
        rowInvAdj[tid] = 1.f / adjSum[rowBase + tid];
        rowInvEud[tid] = 1.f / eudSum[rowBase + tid];
        rowSqn[tid]    = sqn[rowBase + tid];
        for (int q = 0; q < MAXK; q++) { topV[tid][q] = -3.4e38f; topJ[tid][q] = 0; }
    }
    __syncthreads();

    int hi = lane >> 4, nn = lane & 15;
    for (int colBase = 0; colBase < N; colBase += 256) {
        v8f acc[2] = {};
        for (int k0 = 0; k0 < D; k0 += 32) {
            v16bf aF = load_a_frag(F, rowBase, k0, lane, D);
            v16bf b0 = load_b_frag(F, colBase + w * 32,      k0, lane, D);
            v16bf b1 = load_b_frag(F, colBase + w * 32 + 16, k0, lane, D);
            acc[0] = __builtin_amdgcn_wmma_f32_16x16x32_bf16(false, aF, false, b0, (short)0, acc[0], false, false);
            acc[1] = __builtin_amdgcn_wmma_f32_16x16x32_bf16(false, aF, false, b1, (short)0, acc[1], false, false);
        }
        float sqnj[2];
#pragma unroll
        for (int b = 0; b < 2; b++) sqnj[b] = sqn[colBase + w * 32 + b * 16 + nn];
#pragma unroll
        for (int b = 0; b < 2; b++) {
#pragma unroll
            for (int r = 0; r < 8; r++) {
                int il = r + 8 * hi;
                int i  = rowBase + il;
                int j  = colBase + w * 32 + b * 16 + nn;
                float g  = acc[b][r];
                float sq = rowSqn[il] + sqnj[b] - 2.f * g;
                float e  = (sq > 0.f && i != j) ? fast_sqrtf(sq) : 0.f;
                float dist = adj[(size_t)i * N + j] * rowInvAdj[il] + e * rowInvEud[il];
                distLDS[il][w * 32 + b * 16 + nn] = dist;
            }
        }
        __syncthreads();
        if (tid < 16) {   // per-row sorted insertion list (descending)
            for (int cc = 0; cc < 256; cc++) {
                float v = distLDS[tid][cc];
                if (v > topV[tid][kk - 1]) {
                    int pos = kk - 1;
                    while (pos > 0 && topV[tid][pos - 1] < v) {
                        topV[tid][pos] = topV[tid][pos - 1];
                        topJ[tid][pos] = topJ[tid][pos - 1];
                        pos--;
                    }
                    topV[tid][pos] = v;
                    topJ[tid][pos] = colBase + cc;
                }
            }
        }
        __syncthreads();
    }

    if (tid < 16) {
        int i = rowBase + tid;
        size_t base = (size_t)i * kk;
        size_t NK   = (size_t)N * kk;
        for (int q = 0; q < kk; q++) {
            out[base + q]          = (float)i;             // edge_index row 0 (src)
            out[NK + base + q]     = (float)topJ[tid][q];  // edge_index row 1 (dst)
            out[2 * NK + base + q] = topV[tid][q];         // edge_weights
        }
    }
}

extern "C" void kernel_launch(void* const* d_in, const int* in_sizes, int n_in,
                              void* d_out, int out_size, void* d_ws, size_t ws_size,
                              hipStream_t stream) {
    const float* features = (const float*)d_in[0];
    const float* adj      = (const float*)d_in[1];
    // N from adjacency (N*N), D from features, k+1 from output size (no device read needed)
    int N = (int)llround(sqrt((double)in_sizes[1]));
    int D = in_sizes[0] / N;
    int kk = out_size / (3 * N);
    if (kk < 1) kk = 1;
    if (kk > MAXK) kk = MAXK;

    char* ws = (char*)d_ws;
    size_t off = 0;
    bf16_t* fb = (bf16_t*)(ws + off);  off += (size_t)N * D * sizeof(bf16_t); off = (off + 255) & ~(size_t)255;
    float* sqn     = (float*)(ws + off); off += (size_t)N * 4; off = (off + 255) & ~(size_t)255;
    float* adjSum  = (float*)(ws + off); off += (size_t)N * 4; off = (off + 255) & ~(size_t)255;
    int T = N / 128;
    float* eudPart = (float*)(ws + off); off += (size_t)T * N * 4; off = (off + 255) & ~(size_t)255;
    float* eudSum  = (float*)(ws + off); off += (size_t)N * 4;

    hg_prep_features<<<N, 256, 0, stream>>>(features, fb, sqn, D);
    hg_adj_rowsum<<<N, 256, 0, stream>>>(adj, adjSum, N);
    hg_gemm_eud_partial<<<dim3(N / 128, N / 128), 256, 0, stream>>>(fb, sqn, eudPart, N, D);
    hg_eud_reduce<<<(N + 255) / 256, 256, 0, stream>>>(eudPart, eudSum, N, T);
    hg_topk<<<N / 16, 256, 0, stream>>>(fb, sqn, adj, adjSum, eudSum, (float*)d_out, N, D, kk);
}
